// RNAModel_54941221650485
// MI455X (gfx1250) — matrix operand order repaired and checked by reading, not compiled, not run on hardware
//
#include <hip/hip_runtime.h>
#include <hip/hip_bf16.h>

// ---------------------------------------------------------------------------
// MI455X (gfx1250) implementation. All heavy math runs on v_wmma_f32_16x16x32_bf16
// (fp32 accumulate), 2x2 output tiles (32x32) per wave32 for operand reuse and
// back-to-back independent WMMA issue. tri_attn is fused per (b,i) with the
// full 256x256 score matrix resident in 256KB of the 320KB WGP LDS.
// ---------------------------------------------------------------------------

typedef unsigned short u16;
typedef __attribute__((ext_vector_type(16))) __bf16 v16bf;
typedef __attribute__((ext_vector_type(8)))  float  v8f;

constexpr int   Bb  = 2;
constexpr int   Ll  = 256;
constexpr int   Dm  = 256;
constexpr int   Hh  = 4;
constexpr int   FFd = 2048;
constexpr int   Ee  = 64;
constexpr int   ICc = 128;
constexpr int   Nrow = Bb * Ll;            // 512 encoder rows
constexpr long  Pp   = (long)Bb * Ll * Ll; // 131072 pair positions

// ------------------------------- bf16 helpers ------------------------------
__device__ __forceinline__ u16 f2bf(float f) {
  union { float f; unsigned u; } x; x.f = f;
  unsigned r = x.u + 0x7FFFu + ((x.u >> 16) & 1u);   // round-to-nearest-even
  return (u16)(r >> 16);
}
__device__ __forceinline__ float bf2f(u16 h) {
  union { unsigned u; float f; } x; x.u = ((unsigned)h) << 16;
  return x.f;
}

// --------------------------- WMMA operand loaders --------------------------
// A operand (16x32 bf16): lanes 0-15 hold M=lane, K = k0..k0+15;
// lanes 16-31 hold M=lane-16, K = k0+16..k0+31. Packed dword loads.
__device__ __forceinline__ v16bf wmma_load_a(const u16* base, int ld, int m0, int k0, int lane) {
  const u16* p = base + (long)(m0 + (lane & 15)) * ld + k0 + ((lane & 16) ? 16 : 0);
  const unsigned* p0 = (const unsigned*)p;
  const unsigned* p1 = (const unsigned*)(p + 8);
  union { unsigned u[8]; v16bf v; } r;
#pragma unroll
  for (int i = 0; i < 4; ++i) { r.u[i] = p0[i]; r.u[4 + i] = p1[i]; }
  return r.v;
}
// B operand from a row-major weight W[N,K] (B[k,n] = W[n,k]; K contiguous in memory).
// B layout here (contiguous k, same as A): lanes 0-15: N=lane, K=k0..k0+15; lanes 16-31: N=lane-16, K=k0+16..k0+31.
__device__ __forceinline__ v16bf wmma_load_bt(const u16* W, int ld, int n0, int k0, int lane) {
  const unsigned* p = (const unsigned*)(W + (long)(n0 + (lane & 15)) * ld + k0 + ((lane & 16) ? 16 : 0));
  union { unsigned u[8]; v16bf v; } r;
#pragma unroll
  for (int i = 0; i < 8; ++i) r.u[i] = p[i];
  return r.v;
}
// B operand from a row-major matrix Bm[K,N] (K strided) - element gathers.
__device__ __forceinline__ v16bf wmma_load_bn(const u16* Bm, int ld, int n0, int k0, int lane) {
  int n  = n0 + (lane & 15);
  int kb = k0 + ((lane & 16) ? 16 : 0);
  union { u16 h[16]; v16bf v; } r;
#pragma unroll
  for (int i = 0; i < 16; ++i) r.h[i] = Bm[(long)(kb + i) * ld + n];
  return r.v;
}
// A operand sourced from an fp32 LDS tile (convert+pack on the fly).
__device__ __forceinline__ v16bf wmma_load_a_lds(const float* S, int ld, int m0, int k0, int lane) {
  const float* p = S + (long)(m0 + (lane & 15)) * ld + k0 + ((lane & 16) ? 16 : 0);
  union { unsigned u[8]; v16bf v; } r;
#pragma unroll
  for (int i = 0; i < 4; ++i) {
    r.u[i]     = (unsigned)f2bf(p[2 * i])      | ((unsigned)f2bf(p[2 * i + 1])      << 16);
    r.u[4 + i] = (unsigned)f2bf(p[8 + 2 * i]) | ((unsigned)f2bf(p[8 + 2 * i + 1]) << 16);
  }
  return r.v;
}

#define WMMA_BF16(a, b, c) \
  __builtin_amdgcn_wmma_f32_16x16x32_bf16(false, (a), false, (b), (short)0, (c), false, false)

// ----------------------- epilogue + store for one tile ---------------------
// EPI: 0 none, 1 relu, 2 sigmoid, 3 sigmoid(v)*aux, 4 aux*v.
// CHM: bf16 output channel-major [B][N][65536] (plane = L*L).
template<int EPI, bool CHM>
__device__ __forceinline__ void gemm_store(v8f c, int m0, int n0, int lane,
    const float* __restrict__ bias, float scale,
    float* __restrict__ C, long cOff, int ldc,
    u16* __restrict__ Dbf, long dOff, int ldd,
    const float* __restrict__ aux, int ldaux, int N)
{
  const int n  = n0 + (lane & 15);
  const int mb = m0 + ((lane & 16) ? 8 : 0);
  const float bv = bias ? bias[n] : 0.f;
#pragma unroll
  for (int r = 0; r < 8; ++r) {
    const int m = mb + r;
    float v = c[r] * scale + bv;
    if (EPI == 1)      v = v > 0.f ? v : 0.f;
    else if (EPI == 2) v = 1.f / (1.f + __expf(-v));
    else if (EPI == 3) v = (1.f / (1.f + __expf(-v))) * aux[(long)m * ldaux + n];
    else if (EPI == 4) v = aux[(long)m * ldaux + n] * v;
    if (C)   C[cOff + (long)m * ldc + n] = v;
    if (Dbf) {
      if (CHM) Dbf[(((long)(m >> 16) * N + n) << 16) + (m & 65535)] = f2bf(v);
      else     Dbf[dOff + (long)m * ldd + n] = f2bf(v);
    }
  }
}

// ------------------------------ generic GEMM -------------------------------
// C = act( A[M,K](bf16) * B + bias ). One 32x32 output block per wave32:
// 4 independent v_wmma accumulators per K-step (XDL stays busy, operand
// traffic halved vs 16x16/wave). M,N must be multiples of 32; K of 32.
// BMODE 0: B[k,n] = W[n,k] (weight row-major). BMODE 1: B[k,n] = Bm[k,n].
template<int EPI, int BMODE, bool CHM>
__global__ __launch_bounds__(128)
void k_gemm(const u16* __restrict__ A, int lda, long sA, long sA2,
            const u16* __restrict__ Bm, int ldb, long sB, long sB2,
            const float* __restrict__ bias,
            float* __restrict__ C, int ldc, long sC, long sC2,
            u16* __restrict__ Dbf, int ldd, long sD, long sD2,
            const float* __restrict__ aux, int ldaux,
            int M, int N, int K, float scale)
{
  const int wave = threadIdx.x >> 5;
  const int lane = threadIdx.x & 31;
  const int tile = blockIdx.x * (blockDim.x >> 5) + wave;
  const int ntn  = N >> 5;
  if (tile >= (M >> 5) * ntn) return;                  // wave-uniform: EXEC stays all-ones
  const int m0 = (tile / ntn) << 5;
  const int n0 = (tile % ntn) << 5;
  const u16* Ab = A  + blockIdx.y * sA + blockIdx.z * sA2;
  const u16* Bp = Bm + blockIdx.y * sB + blockIdx.z * sB2;
  __builtin_prefetch(Ab + (long)m0 * lda, 0, 1);       // global_prefetch_b8
  v8f c00 = {0.f,0.f,0.f,0.f,0.f,0.f,0.f,0.f};
  v8f c01 = c00, c10 = c00, c11 = c00;
  for (int k = 0; k < K; k += 32) {
    v16bf a0 = wmma_load_a(Ab, lda, m0,      k, lane);
    v16bf a1 = wmma_load_a(Ab, lda, m0 + 16, k, lane);
    v16bf b0, b1;
    if (BMODE == 0) {
      b0 = wmma_load_bt(Bp, ldb, n0,      k, lane);
      b1 = wmma_load_bt(Bp, ldb, n0 + 16, k, lane);
    } else {
      b0 = wmma_load_bn(Bp, ldb, n0,      k, lane);
      b1 = wmma_load_bn(Bp, ldb, n0 + 16, k, lane);
    }
    c00 = WMMA_BF16(a0, b0, c00);
    c01 = WMMA_BF16(a0, b1, c01);
    c10 = WMMA_BF16(a1, b0, c10);
    c11 = WMMA_BF16(a1, b1, c11);
  }
  const long cOff = blockIdx.y * sC + blockIdx.z * sC2;
  const long dOff = blockIdx.y * sD + blockIdx.z * sD2;
  gemm_store<EPI, CHM>(c00, m0,      n0,      lane, bias, scale, C, cOff, ldc, Dbf, dOff, ldd, aux, ldaux, N);
  gemm_store<EPI, CHM>(c01, m0,      n0 + 16, lane, bias, scale, C, cOff, ldc, Dbf, dOff, ldd, aux, ldaux, N);
  gemm_store<EPI, CHM>(c10, m0 + 16, n0,      lane, bias, scale, C, cOff, ldc, Dbf, dOff, ldd, aux, ldaux, N);
  gemm_store<EPI, CHM>(c11, m0 + 16, n0 + 16, lane, bias, scale, C, cOff, ldc, Dbf, dOff, ldd, aux, ldaux, N);
}

// --------------------------- elementwise / LN kernels ----------------------
__global__ void k_f32_to_bf16(const float* __restrict__ in, u16* __restrict__ out, int n) {
  int i = blockIdx.x * 256 + threadIdx.x;
  if (i < n) out[i] = f2bf(in[i]);
}

__global__ void k_embed(const int* __restrict__ tok, const float* __restrict__ emb,
                        float* __restrict__ x, u16* __restrict__ xbf) {
  int i = blockIdx.x * 256 + threadIdx.x;      // Nrow*Dm = 131072
  int row = i >> 8, d = i & 255;
  float v = emb[tok[row] * Dm + d];
  x[i] = v; xbf[i] = f2bf(v);
}

// x = LN(x + r) over D=256; writes fp32 + bf16.
__global__ void k_resln(const float* __restrict__ x, const float* __restrict__ r,
                        const float* __restrict__ g, const float* __restrict__ b,
                        float* __restrict__ xo, u16* __restrict__ xbf) {
  __shared__ float red[256];
  const int row = blockIdx.x, t = threadIdx.x;
  float v = x[row * 256 + t] + r[row * 256 + t];
  red[t] = v; __syncthreads();
  for (int s = 128; s > 0; s >>= 1) { if (t < s) red[t] += red[t + s]; __syncthreads(); }
  float mean = red[0] * (1.f / 256.f); __syncthreads();
  float d = v - mean;
  red[t] = d * d; __syncthreads();
  for (int s = 128; s > 0; s >>= 1) { if (t < s) red[t] += red[t + s]; __syncthreads(); }
  float var = red[0] * (1.f / 256.f);
  float o = d * rsqrtf(var + 1e-5f) * g[t] + b[t];
  xo[row * 256 + t] = o;
  xbf[row * 256 + t] = f2bf(o);
}

// Row LN over last dim C (C==blockDim.x), fp32 in -> bf16 out.
template<int C>
__global__ void k_rowln(const float* __restrict__ in, const float* __restrict__ g,
                        const float* __restrict__ b, u16* __restrict__ out) {
  __shared__ float red[C];
  const long row = blockIdx.x; const int t = threadIdx.x;
  float v = in[row * C + t];
  red[t] = v; __syncthreads();
  for (int s = C / 2; s > 0; s >>= 1) { if (t < s) red[t] += red[t + s]; __syncthreads(); }
  float mean = red[0] * (1.f / C); __syncthreads();
  float d = v - mean;
  red[t] = d * d; __syncthreads();
  for (int s = C / 2; s > 0; s >>= 1) { if (t < s) red[t] += red[t + s]; __syncthreads(); }
  out[row * C + t] = f2bf(d * rsqrtf(red[0] * (1.f / C) + 1e-5f) * g[t] + b[t]);
}

// LN over channels of a channel-major [B][IC][L*L] fp32 tensor -> row-major bf16 [P,IC].
__global__ void k_lnchan(const float* __restrict__ o, const float* __restrict__ g,
                         const float* __restrict__ b, u16* __restrict__ out) {
  __shared__ float red[ICc];
  const long p = blockIdx.x; const int c = threadIdx.x;
  const long bi = p >> 16, off = p & 65535;
  float v = o[((bi * ICc + c) << 16) + off];
  red[c] = v; __syncthreads();
  for (int s = ICc / 2; s > 0; s >>= 1) { if (c < s) red[c] += red[c + s]; __syncthreads(); }
  float mean = red[0] * (1.f / ICc); __syncthreads();
  float d = v - mean;
  red[c] = d * d; __syncthreads();
  for (int s = ICc / 2; s > 0; s >>= 1) { if (c < s) red[c] += red[c + s]; __syncthreads(); }
  out[p * ICc + c] = f2bf(d * rsqrtf(red[0] * (1.f / ICc) + 1e-5f) * g[c] + b[c]);
}

// Encoder attention softmax over last dim (256); fp32 in -> bf16 probabilities.
__global__ void k_softmax_row(const float* __restrict__ s, u16* __restrict__ sbf) {
  __shared__ float red[256];
  const long row = blockIdx.x; const int t = threadIdx.x;
  float v = s[row * 256 + t];
  red[t] = v; __syncthreads();
  for (int st = 128; st > 0; st >>= 1) { if (t < st) red[t] = fmaxf(red[t], red[t + st]); __syncthreads(); }
  float mx = red[0]; __syncthreads();
  float e = __expf(v - mx);
  red[t] = e; __syncthreads();
  for (int st = 128; st > 0; st >>= 1) { if (t < st) red[t] += red[t + st]; __syncthreads(); }
  sbf[row * 256 + t] = f2bf(e / red[0]);
}

__global__ void k_ybuild(const float* __restrict__ pair, const float* __restrict__ eW,
                         const float* __restrict__ eb, float* __restrict__ y) {
  long i = (long)blockIdx.x * 256 + threadIdx.x;   // Pp*Ee
  long p = i >> 6; int e = (int)(i & 63);
  y[i] = pair[p] * eW[e] + eb[e];
}

__global__ void k_bterm(const u16* __restrict__ nbf, const float* __restrict__ Wb,
                        float* __restrict__ bt) {
  long p = (long)blockIdx.x * 256 + threadIdx.x;
  if (p >= Pp) return;
  float s = 0.f;
#pragma unroll
  for (int c = 0; c < Ee; ++c) s += bf2f(nbf[p * Ee + c]) * Wb[c];
  bt[p] = s;
}

__global__ void k_last(const float* __restrict__ y, const float* __restrict__ W,
                       const float* __restrict__ b, float* __restrict__ out) {
  long p = (long)blockIdx.x * 256 + threadIdx.x;
  if (p >= Pp) return;
  float s = b[0];
#pragma unroll
  for (int c = 0; c < Ee; ++c) s += y[p * Ee + c] * W[c];
  out[p] = s;
}

// ------------------- fused triangle attention per (b,i) --------------------
// 256 threads (8 waves); the full 256x256 score matrix lives in 256KB LDS
// (CDNA5 WGP has 320KB). Phase1: S = Q K^T/sqrt(IC)+bterm (WMMA, 32x32/wave).
// Phase2: softmax over the j axis (axis=-2 in the reference == column-wise).
// Phase3: O = g * (A V) (WMMA, A re-read from LDS), bf16 out for the Wo GEMM.
__global__ __launch_bounds__(256)
void k_triattn(const u16* __restrict__ qbf, const u16* __restrict__ kbf,
               const u16* __restrict__ vbf, const float* __restrict__ bt,
               const float* __restrict__ gfp, u16* __restrict__ obf)
{
  extern __shared__ float S[];                 // 256*256 fp32 = 262144 bytes
  const int  bi   = blockIdx.x;                // b*L + i
  const long base = (long)bi * Ll * ICc;
  const u16* Q  = qbf + base;
  const u16* Km = kbf + base;
  const u16* Vm = vbf + base;
  const float* btr = bt + (long)bi * Ll;
  const int wave = threadIdx.x >> 5, lane = threadIdx.x & 31;

  // Phase 1: S = Q K^T * (1/sqrt(IC)) + bterm[j]; 8x8 grid of 32x32 blocks
  for (int t = wave; t < 64; t += 8) {
    const int m0 = (t >> 3) << 5, n0 = (t & 7) << 5;
    v8f c00 = {0.f,0.f,0.f,0.f,0.f,0.f,0.f,0.f};
    v8f c01 = c00, c10 = c00, c11 = c00;
    for (int k = 0; k < ICc; k += 32) {
      v16bf a0 = wmma_load_a(Q, ICc, m0,      k, lane);
      v16bf a1 = wmma_load_a(Q, ICc, m0 + 16, k, lane);
      v16bf b0 = wmma_load_bt(Km, ICc, n0,      k, lane);
      v16bf b1 = wmma_load_bt(Km, ICc, n0 + 16, k, lane);
      c00 = WMMA_BF16(a0, b0, c00);
      c01 = WMMA_BF16(a0, b1, c01);
      c10 = WMMA_BF16(a1, b0, c10);
      c11 = WMMA_BF16(a1, b1, c11);
    }
    const int nl = lane & 15, mh = (lane & 16) ? 8 : 0;
#pragma unroll
    for (int r = 0; r < 8; ++r) {
      const int ma = m0 + mh + r, mb2 = m0 + 16 + mh + r;
      S[ma  * Ll + n0 + nl]      = c00[r] * 0.08838834764831845f + btr[ma];
      S[ma  * Ll + n0 + 16 + nl] = c01[r] * 0.08838834764831845f + btr[ma];
      S[mb2 * Ll + n0 + nl]      = c10[r] * 0.08838834764831845f + btr[mb2];
      S[mb2 * Ll + n0 + 16 + nl] = c11[r] * 0.08838834764831845f + btr[mb2];
    }
  }
  __syncthreads();

  // Phase 2: column softmax (over j)
  {
    const int col = threadIdx.x;
    float mx = -3.4e38f;
    for (int j = 0; j < Ll; ++j) mx = fmaxf(mx, S[j * Ll + col]);
    float sum = 0.f;
    for (int j = 0; j < Ll; ++j) { float e = __expf(S[j * Ll + col] - mx); S[j * Ll + col] = e; sum += e; }
    const float inv = 1.f / sum;
    for (int j = 0; j < Ll; ++j) S[j * Ll + col] *= inv;
  }
  __syncthreads();

  // Phase 3: O = g * (A V); 8x4 grid of 32x32 blocks (M=256, N=IC=128)
  for (int t = wave; t < 32; t += 8) {
    const int m0 = (t >> 2) << 5, n0 = (t & 3) << 5;
    v8f c00 = {0.f,0.f,0.f,0.f,0.f,0.f,0.f,0.f};
    v8f c01 = c00, c10 = c00, c11 = c00;
    for (int k = 0; k < Ll; k += 32) {
      v16bf a0 = wmma_load_a_lds(S, Ll, m0,      k, lane);
      v16bf a1 = wmma_load_a_lds(S, Ll, m0 + 16, k, lane);
      v16bf b0 = wmma_load_bn(Vm, ICc, n0,      k, lane);
      v16bf b1 = wmma_load_bn(Vm, ICc, n0 + 16, k, lane);
      c00 = WMMA_BF16(a0, b0, c00);
      c01 = WMMA_BF16(a0, b1, c01);
      c10 = WMMA_BF16(a1, b0, c10);
      c11 = WMMA_BF16(a1, b1, c11);
    }
    const int nl = lane & 15, mh = (lane & 16) ? 8 : 0;
#pragma unroll
    for (int r = 0; r < 8; ++r) {
      const long ra = (long)bi * Ll + m0 + mh + r;
      const long rb = ra + 16;
      long i00 = ra * ICc + n0 + nl;
      long i01 = ra * ICc + n0 + 16 + nl;
      long i10 = rb * ICc + n0 + nl;
      long i11 = rb * ICc + n0 + 16 + nl;
      obf[i00] = f2bf(c00[r] * gfp[i00]);
      obf[i01] = f2bf(c01[r] * gfp[i01]);
      obf[i10] = f2bf(c10[r] * gfp[i10]);
      obf[i11] = f2bf(c11[r] * gfp[i11]);
    }
  }
}

// --------------------------------- host side -------------------------------
// Input order = jax tree flatten (dict keys sorted): params leaves then tokens.
enum { I_EMB = 0, I_EXPW = 1, I_EXPB = 2, I_LASTW = 3, I_LASTB = 4,
       I_L0 = 5, I_L1 = 21, I_TA = 37, I_TM1 = 47, I_TM2 = 59, I_TM3 = 71, I_TOK = 83 };
enum { EW1 = 0, EW2 = 1, EWK = 2, EWO = 3, EWQ = 4, EWV = 5, EB1 = 6, EB2 = 7,
       EBK = 8, EBO = 9, EBQ = 10, EBV = 11, ELN1B = 12, ELN1G = 13, ELN2B = 14, ELN2G = 15 };
enum { TAWB = 0, TAWG = 1, TAWK = 2, TAWO = 3, TAWQ = 4, TAWV = 5, TABG = 6, TABO = 7, TALNB = 8, TALNG = 9 };
enum { TMW1 = 0, TMW2 = 1, TMWG = 2, TMWO = 3, TMB1 = 4, TMB2 = 5, TMBG = 6, TMBO = 7,
       TMLNB = 8, TMLNG = 9, TMONB = 10, TMONG = 11 };

static inline unsigned gemm_gx(long M, int N) {
  long tiles = (M >> 5) * (N >> 5);              // 32x32 blocks
  return (unsigned)((tiles + 3) >> 2);           // 4 waves (128 threads) per block
}

extern "C" void kernel_launch(void* const* d_in, const int* in_sizes, int n_in,
                              void* d_out, int out_size, void* d_ws, size_t ws_size,
                              hipStream_t stream) {
  (void)in_sizes; (void)n_in; (void)out_size; (void)ws_size;
  char* ws = (char*)d_ws;
  size_t cur = 0;
  auto alloc = [&](size_t bytes) -> char* {
    char* p = ws + cur;
    cur = (cur + bytes + 255) & ~(size_t)255;
    return p;
  };
  auto F = [&](int i) { return (const float*)d_in[i]; };

  // fp32 -> bf16 weight conversion (deterministic layout each call)
  auto cvt = [&](int idx, size_t n) -> const u16* {
    u16* dst = (u16*)alloc(n * 2);
    k_f32_to_bf16<<<dim3((unsigned)((n + 255) / 256)), 256, 0, stream>>>(F(idx), dst, (int)n);
    return dst;
  };

  // encoder weights
  const u16 *eW[2][6];
  for (int l = 0; l < 2; ++l) {
    int b = (l == 0) ? I_L0 : I_L1;
    eW[l][0] = cvt(b + EWQ, Dm * Dm);  eW[l][1] = cvt(b + EWK, Dm * Dm);
    eW[l][2] = cvt(b + EWV, Dm * Dm);  eW[l][3] = cvt(b + EWO, Dm * Dm);
    eW[l][4] = cvt(b + EW1, (size_t)FFd * Dm);
    eW[l][5] = cvt(b + EW2, (size_t)Dm * FFd);
  }
  // triangle-mul weights
  const u16 *tmW[3][4];
  const int tmb[3] = { I_TM1, I_TM2, I_TM3 };
  for (int t = 0; t < 3; ++t) {
    tmW[t][0] = cvt(tmb[t] + TMW1, ICc * Ee);
    tmW[t][1] = cvt(tmb[t] + TMW2, ICc * Ee);
    tmW[t][2] = cvt(tmb[t] + TMWG, Ee * Ee);
    tmW[t][3] = cvt(tmb[t] + TMWO, Ee * ICc);
  }
  // triangle-attention weights
  const u16* taWq = cvt(I_TA + TAWQ, ICc * Ee);
  const u16* taWk = cvt(I_TA + TAWK, ICc * Ee);
  const u16* taWv = cvt(I_TA + TAWV, ICc * Ee);
  const u16* taWg = cvt(I_TA + TAWG, ICc * Ee);
  const u16* taWo = cvt(I_TA + TAWO, Ee * ICc);

  // activations / scratch
  float* XE   = (float*)alloc((size_t)Nrow * Dm * 4);
  u16*   XBF  = (u16*)  alloc((size_t)Nrow * Dm * 2);
  u16*   QBF  = (u16*)  alloc((size_t)Nrow * Dm * 2);
  u16*   KBF  = (u16*)  alloc((size_t)Nrow * Dm * 2);
  u16*   VBF  = (u16*)  alloc((size_t)Nrow * Dm * 2);
  float* SFP  = (float*)alloc((size_t)Bb * Hh * Ll * Ll * 4);
  u16*   SBF  = (u16*)  alloc((size_t)Bb * Hh * Ll * Ll * 2);
  u16*   OBF  = (u16*)  alloc((size_t)Nrow * Dm * 2);
  float* TMPF = (float*)alloc((size_t)Nrow * Dm * 4);
  u16*   FMBF = (u16*)  alloc((size_t)Nrow * FFd * 2);
  float* PAIR = (float*)alloc((size_t)Pp * 4);
  float* Y0   = (float*)alloc((size_t)Pp * Ee * 4);
  float* Y1   = (float*)alloc((size_t)Pp * Ee * 4);
  u16*   NBF  = (u16*)  alloc((size_t)Pp * Ee * 2);
  float* T2F  = (float*)alloc((size_t)Pp * ICc * 4);
  float* GFP  = (float*)alloc((size_t)Pp * ICc * 4);
  u16*   ABF  = (u16*)  alloc((size_t)Pp * ICc * 2);   // chan-major a / ta q
  u16*   KB2  = (u16*)  alloc((size_t)Pp * ICc * 2);
  u16*   VB2  = (u16*)  alloc((size_t)Pp * ICc * 2);
  float* OCH  = (float*)alloc((size_t)Bb * ICc * Ll * Ll * 4);
  u16*   NLN  = (u16*)  alloc((size_t)Pp * ICc * 2);
  float* BT   = (float*)alloc((size_t)Pp * 4);

  // ---------------- embedding ----------------
  k_embed<<<dim3(Nrow * Dm / 256), 256, 0, stream>>>((const int*)d_in[I_TOK], F(I_EMB), XE, XBF);

  // ---------------- encoder layers ----------------
  for (int l = 0; l < 2; ++l) {
    const int pb = (l == 0) ? I_L0 : I_L1;
    // Q/K/V projections (bf16 out only)
    k_gemm<0, 0, false><<<dim3(gemm_gx(Nrow, Dm)), 128, 0, stream>>>(
        XBF, Dm, 0, 0, eW[l][0], Dm, 0, 0, F(pb + EBQ),
        nullptr, 0, 0, 0, QBF, Dm, 0, 0, nullptr, 0, Nrow, Dm, Dm, 1.f);
    k_gemm<0, 0, false><<<dim3(gemm_gx(Nrow, Dm)), 128, 0, stream>>>(
        XBF, Dm, 0, 0, eW[l][1], Dm, 0, 0, F(pb + EBK),
        nullptr, 0, 0, 0, KBF, Dm, 0, 0, nullptr, 0, Nrow, Dm, Dm, 1.f);
    k_gemm<0, 0, false><<<dim3(gemm_gx(Nrow, Dm)), 128, 0, stream>>>(
        XBF, Dm, 0, 0, eW[l][2], Dm, 0, 0, F(pb + EBV),
        nullptr, 0, 0, 0, VBF, Dm, 0, 0, nullptr, 0, Nrow, Dm, Dm, 1.f);
    // S = Q K^T / 8 per (b,h): grid y=h, z=b
    k_gemm<0, 0, false><<<dim3(gemm_gx(Ll, Ll), Hh, Bb), 128, 0, stream>>>(
        QBF, Dm, 64, (long)Ll * Dm, KBF, Dm, 64, (long)Ll * Dm, nullptr,
        SFP, Ll, (long)Ll * Ll, (long)Hh * Ll * Ll, nullptr, 0, 0, 0, nullptr, 0,
        Ll, Ll, 64, 0.125f);
    k_softmax_row<<<dim3(Bb * Hh * Ll), 256, 0, stream>>>(SFP, SBF);
    // O = A V per (b,h)
    k_gemm<0, 1, false><<<dim3(gemm_gx(Ll, 64), Hh, Bb), 128, 0, stream>>>(
        SBF, Ll, (long)Ll * Ll, (long)Hh * Ll * Ll, VBF, Dm, 64, (long)Ll * Dm, nullptr,
        nullptr, 0, 0, 0, OBF, Dm, 64, (long)Ll * Dm, nullptr, 0, Ll, 64, Ll, 1.f);
    // output projection + residual + LN1
    k_gemm<0, 0, false><<<dim3(gemm_gx(Nrow, Dm)), 128, 0, stream>>>(
        OBF, Dm, 0, 0, eW[l][3], Dm, 0, 0, F(pb + EBO),
        TMPF, Dm, 0, 0, nullptr, 0, 0, 0, nullptr, 0, Nrow, Dm, Dm, 1.f);
    k_resln<<<dim3(Nrow), 256, 0, stream>>>(XE, TMPF, F(pb + ELN1G), F(pb + ELN1B), XE, XBF);
    // FFN
    k_gemm<1, 0, false><<<dim3(gemm_gx(Nrow, FFd)), 128, 0, stream>>>(
        XBF, Dm, 0, 0, eW[l][4], Dm, 0, 0, F(pb + EB1),
        nullptr, 0, 0, 0, FMBF, FFd, 0, 0, nullptr, 0, Nrow, FFd, Dm, 1.f);
    k_gemm<0, 0, false><<<dim3(gemm_gx(Nrow, Dm)), 128, 0, stream>>>(
        FMBF, FFd, 0, 0, eW[l][5], FFd, 0, 0, F(pb + EB2),
        TMPF, Dm, 0, 0, nullptr, 0, 0, 0, nullptr, 0, Nrow, Dm, FFd, 1.f);
    k_resln<<<dim3(Nrow), 256, 0, stream>>>(XE, TMPF, F(pb + ELN2G), F(pb + ELN2B), XE, XBF);
  }

  // ---------------- pair = x x^T, expand to E channels ----------------
  k_gemm<0, 0, false><<<dim3(gemm_gx(Ll, Ll), Bb, 1), 128, 0, stream>>>(
      XBF, Dm, (long)Ll * Dm, 0, XBF, Dm, (long)Ll * Dm, 0, nullptr,
      PAIR, Ll, (long)Ll * Ll, 0, nullptr, 0, 0, 0, nullptr, 0, Ll, Ll, Dm, 1.f);
  k_ybuild<<<dim3((unsigned)(Pp * Ee / 256)), 256, 0, stream>>>(PAIR, F(I_EXPW), F(I_EXPB), Y0);

  // ---------------- triangle multiplicative update ----------------
  auto tri_mul = [&](int pb, const u16* const* W, const float* yin, float* yout) {
    k_rowln<Ee><<<dim3((unsigned)Pp), Ee, 0, stream>>>(yin, F(pb + TMLNG), F(pb + TMLNB), NBF);
    // t2 = n W2^T + b2  (fp32)
    k_gemm<0, 0, false><<<dim3(gemm_gx(Pp, ICc)), 128, 0, stream>>>(
        NBF, Ee, 0, 0, W[1], Ee, 0, 0, F(pb + TMB2),
        T2F, ICc, 0, 0, nullptr, 0, 0, 0, nullptr, 0, (int)Pp, ICc, Ee, 1.f);
    // a = sigmoid(n W1^T + b1) * t2  -> bf16 channel-major [B][IC][L*L]
    k_gemm<3, 0, true><<<dim3(gemm_gx(Pp, ICc)), 128, 0, stream>>>(
        NBF, Ee, 0, 0, W[0], Ee, 0, 0, F(pb + TMB1),
        nullptr, 0, 0, 0, ABF, 0, 0, 0, T2F, ICc, (int)Pp, ICc, Ee, 1.f);
    // g = sigmoid(n Wg^T + bg)  [P,E] fp32
    k_gemm<2, 0, false><<<dim3(gemm_gx(Pp, Ee)), 128, 0, stream>>>(
        NBF, Ee, 0, 0, W[2], Ee, 0, 0, F(pb + TMBG),
        GFP, Ee, 0, 0, nullptr, 0, 0, 0, nullptr, 0, (int)Pp, Ee, Ee, 1.f);
    // o_c = A_c A_c^T, batched over (b,c) = 256 GEMMs of 256^3
    k_gemm<0, 0, false><<<dim3(gemm_gx(Ll, Ll), Bb * ICc, 1), 128, 0, stream>>>(
        ABF, Ll, (long)Ll * Ll, 0, ABF, Ll, (long)Ll * Ll, 0, nullptr,
        OCH, Ll, (long)Ll * Ll, 0, nullptr, 0, 0, 0, nullptr, 0, Ll, Ll, Ll, 1.f);
    k_lnchan<<<dim3((unsigned)Pp), ICc, 0, stream>>>(OCH, F(pb + TMONG), F(pb + TMONB), NLN);
    // y = g * (ln(o) Wo^T + bo)
    k_gemm<4, 0, false><<<dim3(gemm_gx(Pp, Ee)), 128, 0, stream>>>(
        NLN, ICc, 0, 0, W[3], ICc, 0, 0, F(pb + TMBO),
        yout, Ee, 0, 0, nullptr, 0, 0, 0, GFP, Ee, (int)Pp, Ee, ICc, 1.f);
  };

  tri_mul(I_TM1, tmW[0], Y0, Y1);

  // ---------------- triangle attention ----------------
  {
    const int pb = I_TA;
    k_rowln<Ee><<<dim3((unsigned)Pp), Ee, 0, stream>>>(Y1, F(pb + TALNG), F(pb + TALNB), NBF);
    k_gemm<0, 0, false><<<dim3(gemm_gx(Pp, ICc)), 128, 0, stream>>>(
        NBF, Ee, 0, 0, taWq, Ee, 0, 0, nullptr,
        nullptr, 0, 0, 0, ABF, ICc, 0, 0, nullptr, 0, (int)Pp, ICc, Ee, 1.f);
    k_gemm<0, 0, false><<<dim3(gemm_gx(Pp, ICc)), 128, 0, stream>>>(
        NBF, Ee, 0, 0, taWk, Ee, 0, 0, nullptr,
        nullptr, 0, 0, 0, KB2, ICc, 0, 0, nullptr, 0, (int)Pp, ICc, Ee, 1.f);
    k_gemm<0, 0, false><<<dim3(gemm_gx(Pp, ICc)), 128, 0, stream>>>(
        NBF, Ee, 0, 0, taWv, Ee, 0, 0, nullptr,
        nullptr, 0, 0, 0, VB2, ICc, 0, 0, nullptr, 0, (int)Pp, ICc, Ee, 1.f);
    k_gemm<2, 0, false><<<dim3(gemm_gx(Pp, ICc)), 128, 0, stream>>>(
        NBF, Ee, 0, 0, taWg, Ee, 0, 0, F(pb + TABG),
        GFP, ICc, 0, 0, nullptr, 0, 0, 0, nullptr, 0, (int)Pp, ICc, Ee, 1.f);
    k_bterm<<<dim3((unsigned)((Pp + 255) / 256)), 256, 0, stream>>>(NBF, F(pb + TAWB), BT);
    // fused attention per (b,i): 512 blocks x 256 threads, 256KB dynamic LDS
    k_triattn<<<dim3(Bb * Ll), 256, Ll * Ll * sizeof(float), stream>>>(ABF, KB2, VB2, BT, GFP, NLN);
    k_gemm<0, 0, false><<<dim3(gemm_gx(Pp, Ee)), 128, 0, stream>>>(
        NLN, ICc, 0, 0, taWo, ICc, 0, 0, F(pb + TABO),
        Y0, Ee, 0, 0, nullptr, 0, 0, 0, nullptr, 0, (int)Pp, Ee, ICc, 1.f);
  }

  tri_mul(I_TM2, tmW[1], Y0, Y1);
  tri_mul(I_TM3, tmW[2], Y1, Y0);

  // ---------------- final projection E -> 1 ----------------
  k_last<<<dim3((unsigned)((Pp + 255) / 256)), 256, 0, stream>>>(Y0, F(I_LASTW), F(I_LASTB), (float*)d_out);
}
